// CausalSelfAttention_10428180594716
// MI455X (gfx1250) — matrix operand order repaired and checked by reading, not compiled, not run on hardware
//
#include <hip/hip_runtime.h>
#include <hip/hip_bf16.h>
#include <stdint.h>

typedef __attribute__((ext_vector_type(16))) __bf16 v16bf;
typedef __attribute__((ext_vector_type(8)))  float  v8f;
typedef unsigned short u16;
typedef unsigned int   u32;
typedef unsigned long long u64;

static constexpr int Bc = 2, Tc = 2048, Dm = 2048, Hn = 16, Dh = 128;
static constexpr int Mrows = Bc * Tc; // 4096
#define ATT_SCALE 0.08838834764831845f  /* 128^-0.5 */
#define CAP 50.0f

__device__ __forceinline__ u16 f2bf(float f) {
    union { float f; u32 u; } x; x.f = f;
    u32 r = x.u + 0x7FFFu + ((x.u >> 16) & 1u);   // round-to-nearest-even
    return (u16)(r >> 16);
}

union FragB { uint4 q[2]; v16bf v; };

// low 32 bits of the flat address of an LDS location == LDS byte offset
__device__ __forceinline__ u32 lds_off(const void* p) {
    return (u32)(u64)(uintptr_t)p;
}
// CDNA5 async copy: global -> LDS, tracked by ASYNCcnt (no VGPR data path)
__device__ __forceinline__ void async_ld_b128(u32 ldsoff, const void* gptr) {
    asm volatile("global_load_async_to_lds_b128 %0, %1, off"
                 :: "v"(ldsoff), "v"((u64)(uintptr_t)gptr) : "memory");
}
__device__ __forceinline__ void wait_async0() {
    asm volatile("s_wait_asynccnt 0" ::: "memory");
}

// ---------------------------------------------------------------- f32 -> bf16
__global__ __launch_bounds__(256) void k_cvt(const float* __restrict__ in,
                                             u16* __restrict__ out, int n4) {
    int i = blockIdx.x * 256 + threadIdx.x;
    if (i < n4) {
        float4 v = reinterpret_cast<const float4*>(in)[i];
        u32 lo = (u32)f2bf(v.x) | ((u32)f2bf(v.y) << 16);
        u32 hi = (u32)f2bf(v.z) | ((u32)f2bf(v.w) << 16);
        reinterpret_cast<uint2*>(out)[i] = make_uint2(lo, hi);
    }
}

// ------------------------------------------------- bf16 WMMA GEMM: out = A*W^T
// A: [M,K] bf16 row-major.  W: [N,K] bf16 row-major.
// mode 0: outF[M,N] f32 row-major.   mode 1: outB bf16 permuted to [b,h,t,d].
// Double-buffered LDS, async global->LDS copies, 1 barrier per K-step.
#define LP 40   /* padded tile row (halfs): 80B stride -> conflict-free frags */
__global__ __launch_bounds__(256) void k_gemm(const u16* __restrict__ A,
                                              const u16* __restrict__ W,
                                              float* __restrict__ outF,
                                              u16* __restrict__ outB,
                                              int M, int N, int K, int mode) {
    __shared__ u16 Al[2][128][LP];
    __shared__ u16 Bl[2][128][LP];
    const int tid  = threadIdx.x;
    const int wave = tid >> 5, lane = tid & 31;
    const int g = lane >> 4, ln = lane & 15;
    const int m0 = blockIdx.y * 128, n0 = blockIdx.x * 128;
    const int wr = wave >> 1, wc = wave & 1;   // 4x2 wave grid: 32x64 per wave

    // per-thread staging slots (2 b128 per tile per matrix)
    const int r0 = tid >> 2,          c0 = (tid & 3) * 8;
    const int r1 = (tid + 256) >> 2,  c1 = ((tid + 256) & 3) * 8;
    const u16* gA0 = A + (size_t)(m0 + r0) * K + c0;
    const u16* gA1 = A + (size_t)(m0 + r1) * K + c1;
    const u16* gW0 = W + (size_t)(n0 + r0) * K + c0;
    const u16* gW1 = W + (size_t)(n0 + r1) * K + c1;
    u32 lA[2][2], lB[2][2];
#pragma unroll
    for (int b = 0; b < 2; b++) {
        lA[b][0] = lds_off(&Al[b][r0][c0]);
        lA[b][1] = lds_off(&Al[b][r1][c1]);
        lB[b][0] = lds_off(&Bl[b][r0][c0]);
        lB[b][1] = lds_off(&Bl[b][r1][c1]);
    }

    const v8f zero = {0.f,0.f,0.f,0.f,0.f,0.f,0.f,0.f};
    v8f acc[2][4];
#pragma unroll
    for (int i = 0; i < 2; i++)
#pragma unroll
        for (int j = 0; j < 4; j++) acc[i][j] = zero;

    // prologue: tile 0 -> buffer 0
    async_ld_b128(lA[0][0], gA0);
    async_ld_b128(lA[0][1], gA1);
    async_ld_b128(lB[0][0], gW0);
    async_ld_b128(lB[0][1], gW1);

    const int niter = K >> 5;
    for (int it = 0; it < niter; ++it) {
        wait_async0();        // this wave's tile-`it` copies landed in LDS
        __syncthreads();      // all waves: tile ready; prev reads of nxt done
        const int cur = it & 1, nxt = cur ^ 1;
        const int kbn = (it + 1) << 5;
        if (kbn < K) {        // overlap tile it+1 copies with tile it compute
            async_ld_b128(lA[nxt][0], gA0 + kbn);
            async_ld_b128(lA[nxt][1], gA1 + kbn);
            async_ld_b128(lB[nxt][0], gW0 + kbn);
            async_ld_b128(lB[nxt][1], gW1 + kbn);
        }

        FragB af[2], bf[4];
#pragma unroll
        for (int i = 0; i < 2; i++) {
            int r = wr * 32 + i * 16 + ln;     // A rows: lanes 0-15 == 16-31
            af[i].q[0] = *reinterpret_cast<const uint4*>(&Al[cur][r][g * 8]);
            af[i].q[1] = *reinterpret_cast<const uint4*>(&Al[cur][r][16 + g * 8]);
        }
#pragma unroll
        for (int j = 0; j < 4; j++) {
            int c = wc * 64 + j * 16 + ln;     // B column = lane; k in table order
            bf[j].q[0] = *reinterpret_cast<const uint4*>(&Bl[cur][c][g * 8]);
            bf[j].q[1] = *reinterpret_cast<const uint4*>(&Bl[cur][c][16 + g * 8]);
        }
#pragma unroll
        for (int i = 0; i < 2; i++)
#pragma unroll
            for (int j = 0; j < 4; j++)
                acc[i][j] = __builtin_amdgcn_wmma_f32_16x16x32_bf16(
                    false, af[i].v, false, bf[j].v, (short)0, acc[i][j], false, false);
    }

#pragma unroll
    for (int i = 0; i < 2; i++)
#pragma unroll
        for (int j = 0; j < 4; j++)
#pragma unroll
            for (int r = 0; r < 8; r++) {
                int row = m0 + wr * 32 + i * 16 + g * 8 + r;
                int col = n0 + wc * 64 + j * 16 + ln;
                float v = acc[i][j][r];
                if (mode == 1) {
                    int b = row >> 11, t = row & (Tc - 1);
                    int h = col >> 7,  d = col & (Dh - 1);
                    outB[(((size_t)(b * Hn + h)) * Tc + t) * Dh + d] = f2bf(v);
                } else {
                    outF[(size_t)row * N + col] = v;
                }
            }
}

// ------------------------------------- per-head RMSNorm + RoPE, f32 -> bf16
// src: f32 [b,t,h*128]; dst: bf16 [b,h,t,d]
__global__ __launch_bounds__(256) void k_norm_rope(const float* __restrict__ src,
                                                   const float* __restrict__ w,
                                                   const float* __restrict__ cs,
                                                   const float* __restrict__ sn,
                                                   u16* __restrict__ dst) {
    int wave = threadIdx.x >> 5, lane = threadIdx.x & 31;
    int rowid = blockIdx.x * 8 + wave;          // (b,t,h) flat
    int b = rowid / (Tc * Hn);
    int rem = rowid - b * Tc * Hn;
    int t = rem / Hn, h = rem - t * Hn;
    const float* p = src + ((size_t)(b * Tc + t) * Dm + h * Dh);
    float4 x = reinterpret_cast<const float4*>(p)[lane];
    float ss = x.x * x.x + x.y * x.y + x.z * x.z + x.w * x.w;
#pragma unroll
    for (int m = 1; m < 32; m <<= 1) ss += __shfl_xor(ss, m, 32);
    float rn = rsqrtf(ss * (1.0f / 128.0f) + 1e-6f);
    float4 wv = reinterpret_cast<const float4*>(w)[lane];
    float y0 = x.x * rn * wv.x, y1 = x.y * rn * wv.y;
    float y2 = x.z * rn * wv.z, y3 = x.w * rn * wv.w;
    int i0 = lane * 2, i1 = lane * 2 + 1;
    float c0 = cs[t * 64 + i0], s0 = sn[t * 64 + i0];
    float c1 = cs[t * 64 + i1], s1 = sn[t * 64 + i1];
    float o0 = y0 * c0 - y1 * s0, o1 = y0 * s0 + y1 * c0;
    float o2 = y2 * c1 - y3 * s1, o3 = y2 * s1 + y3 * c1;
    u16* q = dst + (((size_t)(b * Hn + h)) * Tc + t) * Dh + lane * 4;
    *reinterpret_cast<uint2*>(q) =
        make_uint2((u32)f2bf(o0) | ((u32)f2bf(o1) << 16),
                   (u32)f2bf(o2) | ((u32)f2bf(o3) << 16));
}

// --------------------------------------------- flash attention, softcap+causal
// Q,K,V: bf16 [b,h,t,128].  O: bf16 [b*t, h*128]  (rows for final GEMM)
#define KLP 144  /* K-tile padded row (halfs), 288B stride, 16B aligned */
__global__ __launch_bounds__(128) void k_attn(const u16* __restrict__ Q,
                                              const u16* __restrict__ Kk,
                                              const u16* __restrict__ V,
                                              u16* __restrict__ O) {
    __shared__ u16 Kl[32][KLP];     // K tile [key][d]
    __shared__ u16 Vt[128][40];     // V tile transposed [d][key]
    __shared__ u16 Pl[4][16][40];   // per-wave P staging
    const int tid  = threadIdx.x;
    const int wave = tid >> 5, lane = tid & 31;
    const int g = lane >> 4, ln = lane & 15;
    const int bh = blockIdx.y;
    const int b = bh >> 4, h = bh & 15;
    const int q0 = blockIdx.x * 64;
    const int qrow0 = q0 + wave * 16;
    const u16* Qbh = Q  + (size_t)bh * Tc * Dh;
    const u16* Kbh = Kk + (size_t)bh * Tc * Dh;
    const u16* Vbh = V  + (size_t)bh * Tc * Dh;

    // per-thread async K staging slots (4 b128 each)
    u32 klds[4];
    int krow[4], kch[4];
#pragma unroll
    for (int j = 0; j < 4; j++) {
        int u = tid + 128 * j;             // 512 b128 slots per tile
        krow[j] = u >> 4; kch[j] = (u & 15) * 8;
        klds[j] = lds_off(&Kl[krow[j]][kch[j]]);
    }

    FragB qf[4];
#pragma unroll
    for (int c = 0; c < 4; c++) {
        const u16* qp = Qbh + (size_t)(qrow0 + ln) * Dh + c * 32;
        qf[c].q[0] = *reinterpret_cast<const uint4*>(qp + g * 8);
        qf[c].q[1] = *reinterpret_cast<const uint4*>(qp + 16 + g * 8);
    }

    const v8f zero = {0.f,0.f,0.f,0.f,0.f,0.f,0.f,0.f};
    v8f o[8];
    float mrow[8], lrow[8];
#pragma unroll
    for (int c = 0; c < 8; c++) o[c] = zero;
#pragma unroll
    for (int r = 0; r < 8; r++) { mrow[r] = -1e30f; lrow[r] = 0.f; }

    const int nkt = q0 / 32 + 2;   // uniform per block; mask handles the edge
    for (int kt = 0; kt < nkt; ++kt) {
        __syncthreads();           // all waves done reading Kl/Vt of prev tile
#pragma unroll
        for (int j = 0; j < 4; j++)                      // async K -> LDS
            async_ld_b128(klds[j],
                          Kbh + (size_t)(kt * 32 + krow[j]) * Dh + kch[j]);
        {                                                // stage V transposed
            int key = tid >> 2;
            int dbase = (tid & 3) * 32;
#pragma unroll
            for (int j = 0; j < 16; j++) {
                u32 pv = *reinterpret_cast<const u32*>(
                    &Vbh[(size_t)(kt * 32 + key) * Dh + dbase + j * 2]);
                Vt[dbase + j * 2][key]     = (u16)(pv & 0xFFFFu);
                Vt[dbase + j * 2 + 1][key] = (u16)(pv >> 16);
            }
        }
        wait_async0();
        __syncthreads();

        v8f s0 = zero, s1 = zero;                        // scores: Q x K^T
#pragma unroll
        for (int c = 0; c < 4; c++) {
            FragB b0, b1;
            b0.q[0] = *reinterpret_cast<const uint4*>(&Kl[ln][c * 32 + g * 8]);
            b0.q[1] = *reinterpret_cast<const uint4*>(&Kl[ln][c * 32 + 16 + g * 8]);
            b1.q[0] = *reinterpret_cast<const uint4*>(&Kl[16 + ln][c * 32 + g * 8]);
            b1.q[1] = *reinterpret_cast<const uint4*>(&Kl[16 + ln][c * 32 + 16 + g * 8]);
            s0 = __builtin_amdgcn_wmma_f32_16x16x32_bf16(
                false, qf[c].v, false, b0.v, (short)0, s0, false, false);
            s1 = __builtin_amdgcn_wmma_f32_16x16x32_bf16(
                false, qf[c].v, false, b1.v, (short)0, s1, false, false);
        }

        float p0[8], p1[8];
        const int kc0 = kt * 32 + ln, kc1 = kt * 32 + 16 + ln;
#pragma unroll
        for (int r = 0; r < 8; r++) {                    // softcap+mask+softmax
            int qr = qrow0 + g * 8 + r;
            float a = s0[r] * ATT_SCALE, bb = s1[r] * ATT_SCALE;
            a  = CAP * tanhf(a  * (1.0f / CAP));
            bb = CAP * tanhf(bb * (1.0f / CAP));
            if (kc0 > qr) a  = -1e30f;
            if (kc1 > qr) bb = -1e30f;
            float mx = fmaxf(a, bb);
#pragma unroll
            for (int m = 1; m < 16; m <<= 1) mx = fmaxf(mx, __shfl_xor(mx, m, 32));
            float mnew = fmaxf(mrow[r], mx);
            float corr = __expf(mrow[r] - mnew);
            float e0 = __expf(a - mnew), e1 = __expf(bb - mnew);
            float sm = e0 + e1;
#pragma unroll
            for (int m = 1; m < 16; m <<= 1) sm += __shfl_xor(sm, m, 32);
            lrow[r] = lrow[r] * corr + sm;
            mrow[r] = mnew;
            p0[r] = e0; p1[r] = e1;
#pragma unroll
            for (int c = 0; c < 8; c++) o[c][r] *= corr;
        }

#pragma unroll
        for (int r = 0; r < 8; r++) {                    // D-layout -> A-layout
            Pl[wave][g * 8 + r][ln]      = f2bf(p0[r]);
            Pl[wave][g * 8 + r][16 + ln] = f2bf(p1[r]);
        }
        asm volatile("s_wait_dscnt 0" ::: "memory");
        FragB pa;
        pa.q[0] = *reinterpret_cast<const uint4*>(&Pl[wave][ln][g * 8]);
        pa.q[1] = *reinterpret_cast<const uint4*>(&Pl[wave][ln][16 + g * 8]);
#pragma unroll
        for (int c = 0; c < 8; c++) {                    // O += P x V
            FragB bv;
            bv.q[0] = *reinterpret_cast<const uint4*>(&Vt[c * 16 + ln][g * 8]);
            bv.q[1] = *reinterpret_cast<const uint4*>(&Vt[c * 16 + ln][16 + g * 8]);
            o[c] = __builtin_amdgcn_wmma_f32_16x16x32_bf16(
                false, pa.v, false, bv.v, (short)0, o[c], false, false);
        }
    }

#pragma unroll
    for (int c = 0; c < 8; c++)
#pragma unroll
        for (int r = 0; r < 8; r++) {
            int qr = qrow0 + g * 8 + r;
            int d = c * 16 + ln;
            float v = o[c][r] / lrow[r];
            O[((size_t)(b * Tc + qr)) * Dm + h * Dh + d] = f2bf(v);
        }
}

// -----------------------------------------------------------------------------
extern "C" void kernel_launch(void* const* d_in, const int* in_sizes, int n_in,
                              void* d_out, int out_size, void* d_ws, size_t ws_size,
                              hipStream_t stream) {
    (void)in_sizes; (void)n_in; (void)out_size; (void)ws_size;
    const float* x  = (const float*)d_in[0];
    const float* wq = (const float*)d_in[1];
    const float* wk = (const float*)d_in[2];
    const float* wv = (const float*)d_in[3];
    const float* wo = (const float*)d_in[4];
    const float* qn = (const float*)d_in[5];
    const float* kn = (const float*)d_in[6];
    const float* rc = (const float*)d_in[7];
    const float* rs = (const float*)d_in[8];

    char* ws = (char*)d_ws;
    size_t off = 0;
    auto alloc = [&](size_t bytes) -> char* {
        char* p = ws + off; off += (bytes + 255) & ~(size_t)255; return p;
    };
    const size_t nX = (size_t)Mrows * Dm;   // 8M elements
    const size_t nW = (size_t)Dm * Dm;      // 4M elements
    u16*   xb  = (u16*)alloc(nX * 2);
    u16*   wqb = (u16*)alloc(nW * 2);
    u16*   wkb = (u16*)alloc(nW * 2);
    u16*   wvb = (u16*)alloc(nW * 2);
    u16*   wob = (u16*)alloc(nW * 2);
    float* qf  = (float*)alloc(nX * 4);
    float* kf  = (float*)alloc(nX * 4);
    u16*   qb  = (u16*)alloc(nX * 2);
    u16*   kb  = (u16*)alloc(nX * 2);
    u16*   vb  = (u16*)alloc(nX * 2);
    u16*   ao  = (u16*)alloc(nX * 2);

    k_cvt<<<dim3((unsigned)(nX / 4 / 256)), 256, 0, stream>>>(x,  xb,  (int)(nX / 4));
    k_cvt<<<dim3((unsigned)(nW / 4 / 256)), 256, 0, stream>>>(wq, wqb, (int)(nW / 4));
    k_cvt<<<dim3((unsigned)(nW / 4 / 256)), 256, 0, stream>>>(wk, wkb, (int)(nW / 4));
    k_cvt<<<dim3((unsigned)(nW / 4 / 256)), 256, 0, stream>>>(wv, wvb, (int)(nW / 4));
    k_cvt<<<dim3((unsigned)(nW / 4 / 256)), 256, 0, stream>>>(wo, wob, (int)(nW / 4));

    dim3 gg(Dm / 128, Mrows / 128);   // (16, 32)
    k_gemm<<<gg, 256, 0, stream>>>(xb, wqb, qf, nullptr, Mrows, Dm, Dm, 0);
    k_gemm<<<gg, 256, 0, stream>>>(xb, wkb, kf, nullptr, Mrows, Dm, Dm, 0);
    k_gemm<<<gg, 256, 0, stream>>>(xb, wvb, nullptr, vb, Mrows, Dm, Dm, 1);

    k_norm_rope<<<dim3(Bc * Tc * Hn / 8), 256, 0, stream>>>(qf, qn, rc, rs, qb);
    k_norm_rope<<<dim3(Bc * Tc * Hn / 8), 256, 0, stream>>>(kf, kn, rc, rs, kb);

    k_attn<<<dim3(Tc / 64, Bc * Hn), 128, 0, stream>>>(qb, kb, vb, ao);

    k_gemm<<<gg, 256, 0, stream>>>(ao, wob, (float*)d_out, nullptr, Mrows, Dm, Dm, 0);
}
